// NextHopGNN_37288906064565
// MI455X (gfx1250) — compile-verified, matrix operand order checked by reading, not run on hardware
//
#include <hip/hip_runtime.h>

typedef __attribute__((ext_vector_type(16))) __bf16 v16bf;
typedef __attribute__((ext_vector_type(8)))  float  v8f;

// ---------------------------------------------------------------------------
// Degree / normalization kernels
// ---------------------------------------------------------------------------
__global__ void init_deg_kernel(float* __restrict__ deg, int n) {
  int i = blockIdx.x * blockDim.x + threadIdx.x;
  if (i < n) deg[i] = 1.0f;  // self-loop contributes 1 to every node's degree
}

__global__ void count_deg_kernel(const int* __restrict__ dst, float* __restrict__ deg, int e) {
  int i = blockIdx.x * blockDim.x + threadIdx.x;
  if (i < e) atomicAdd(&deg[dst[i]], 1.0f);
}

__global__ void dinv_kernel(float* __restrict__ deg, int n) {
  int i = blockIdx.x * blockDim.x + threadIdx.x;
  if (i < n) {
    float d = deg[i];
    deg[i] = d > 0.0f ? rsqrtf(d) : 0.0f;  // in place: deg -> dinv
  }
}

// ---------------------------------------------------------------------------
// Pre-pack W[K x 64] (f32, row-major) into bf16 laid out exactly as the WMMA
// B operand registers: slot = (ktIdx*4 + nt)*32 + lane, 16 bf16 per slot.
// Lane packing (B^T convention): col n = nt*16 + (lane&15),
//   elems 0..7  = K (ktIdx*32 + khalf + 0..7),
//   elems 8..15 = K (ktIdx*32 + khalf + 16..23),  khalf = (lane>>4)*8.
// ---------------------------------------------------------------------------
template <int K>
__global__ void pack_W_kernel(const float* __restrict__ W, __bf16* __restrict__ Bp) {
  constexpr int slots = (K / 32) * 4 * 32;
  int t = blockIdx.x * blockDim.x + threadIdx.x;
  if (t >= slots) return;
  int lane  = t & 31;
  int nt    = (t >> 5) & 3;
  int ktIdx = t >> 7;
  int r16   = lane & 15;
  int khalf = (lane >> 4) * 8;
  int col   = nt * 16 + r16;
  v16bf b;
#pragma unroll
  for (int i = 0; i < 8; ++i) {
    b[i]     = (__bf16)W[(size_t)(ktIdx * 32 + khalf + i) * 64 + col];
    b[8 + i] = (__bf16)W[(size_t)(ktIdx * 32 + khalf + 16 + i) * 64 + col];
  }
  *(v16bf*)(Bp + (size_t)t * 16) = b;
}

// ---------------------------------------------------------------------------
// WMMA GEMM: out[nRows x 64] = f(A[nRows x K]) @ W[K x 64]
// f = relu(x + bias) when RELU_BIAS, else identity.
// One wave computes a 16-row x 64-col tile: 4 accumulators, K-loop step 32.
// A: f32 from memory, two contiguous float4-pairs per K-step, cvt to bf16.
// B: one 32-byte (2x b128) load per tile from the pre-packed bf16 buffer.
// ---------------------------------------------------------------------------
template <int K, bool RELU_BIAS>
__global__ __launch_bounds__(256) void gemm_wmma_kernel(
    const float* __restrict__ A, const __bf16* __restrict__ Bp,
    const float* __restrict__ bias, float* __restrict__ out, int nRows) {
  const int lane = threadIdx.x & 31;
  const int wave = threadIdx.x >> 5;
  const int rowTile = (blockIdx.x * 8 + wave) * 16;
  if (rowTile >= nRows) return;  // uniform per wave -> EXEC stays all-1s

  const int r16   = lane & 15;        // M row within tile
  const int khalf = (lane >> 4) * 8;  // 0 or 8: K sub-half selector
  int row  = rowTile + r16;
  int rowc = row < nRows ? row : nRows - 1;
  const float* __restrict__ arow = A + (size_t)rowc * K;

  v8f acc[4];
  acc[0] = (v8f){}; acc[1] = (v8f){}; acc[2] = (v8f){}; acc[3] = (v8f){};

#pragma unroll
  for (int ktIdx = 0; ktIdx < K / 32; ++ktIdx) {
    const int kt = ktIdx * 32;
    // ---- A fragment: two 8-float contiguous runs -> 4x b128 loads ----
    const float4 a0 = *(const float4*)(arow + kt + khalf);
    const float4 a1 = *(const float4*)(arow + kt + khalf + 4);
    const float4 a2 = *(const float4*)(arow + kt + khalf + 16);
    const float4 a3 = *(const float4*)(arow + kt + khalf + 20);
    float lo[8] = {a0.x, a0.y, a0.z, a0.w, a1.x, a1.y, a1.z, a1.w};
    float hi[8] = {a2.x, a2.y, a2.z, a2.w, a3.x, a3.y, a3.z, a3.w};
    if constexpr (RELU_BIAS) {
      const float4 bb0 = *(const float4*)(bias + kt + khalf);
      const float4 bb1 = *(const float4*)(bias + kt + khalf + 4);
      const float4 bb2 = *(const float4*)(bias + kt + khalf + 16);
      const float4 bb3 = *(const float4*)(bias + kt + khalf + 20);
      float blo[8] = {bb0.x, bb0.y, bb0.z, bb0.w, bb1.x, bb1.y, bb1.z, bb1.w};
      float bhi[8] = {bb2.x, bb2.y, bb2.z, bb2.w, bb3.x, bb3.y, bb3.z, bb3.w};
#pragma unroll
      for (int i = 0; i < 8; ++i) {
        lo[i] += blo[i]; lo[i] = lo[i] > 0.0f ? lo[i] : 0.0f;
        hi[i] += bhi[i]; hi[i] = hi[i] > 0.0f ? hi[i] : 0.0f;
      }
    }
    v16bf a;
#pragma unroll
    for (int i = 0; i < 8; ++i) {
      a[i]     = (__bf16)lo[i];
      a[8 + i] = (__bf16)hi[i];
    }
    // ---- 4 pre-packed B fragments + WMMA ----
#pragma unroll
    for (int nt = 0; nt < 4; ++nt) {
      const v16bf b =
          *(const v16bf*)(Bp + ((size_t)(ktIdx * 4 + nt) * 32 + lane) * 16);
      acc[nt] = __builtin_amdgcn_wmma_f32_16x16x32_bf16(
          /*neg_a=*/false, a, /*neg_b=*/false, b,
          /*c_mod=*/(short)0, acc[nt], /*reuse_a=*/false, /*reuse_b=*/false);
    }
  }

  // ---- Store D: vgpr r holds row (rowTile + r + 8*(lane>=16)), col nt*16+r16
  const int rbase = rowTile + ((lane >> 4) * 8);
#pragma unroll
  for (int nt = 0; nt < 4; ++nt) {
    const int col = nt * 16 + r16;
#pragma unroll
    for (int r = 0; r < 8; ++r) {
      int rr = rbase + r;
      if (rr < nRows) out[(size_t)rr * 64 + col] = acc[nt][r];
    }
  }
}

// ---------------------------------------------------------------------------
// out[i, :] = x[i, :] * dinv[i]^2 (+ bias)   -- self-loop term, full overwrite
// 16 threads per node, float4 per thread.
// ---------------------------------------------------------------------------
__global__ void selfloop_init_kernel(const float* __restrict__ x,
                                     const float* __restrict__ dinv,
                                     const float* __restrict__ bias,
                                     float* __restrict__ out, int n) {
  int t = blockIdx.x * blockDim.x + threadIdx.x;
  int i = t >> 4;
  if (i >= n) return;
  int f = (t & 15) * 4;
  float di  = dinv[i];
  float nrm = di * di;
  const float4 v = *(const float4*)(x + (size_t)i * 64 + f);
  float4 o;
  o.x = v.x * nrm; o.y = v.y * nrm; o.z = v.z * nrm; o.w = v.w * nrm;
  if (bias != nullptr) {
    o.x += bias[f]; o.y += bias[f + 1]; o.z += bias[f + 2]; o.w += bias[f + 3];
  }
  *(float4*)(out + (size_t)i * 64 + f) = o;
}

// ---------------------------------------------------------------------------
// Per-edge scatter: out[dst, :] += x[src, :] * dinv[src] * dinv[dst]
// 16 threads per edge, float4 gather + 4x global_atomic_add_f32.
// ---------------------------------------------------------------------------
__global__ void scatter_edges_kernel(const float* __restrict__ x,
                                     const float* __restrict__ dinv,
                                     const int* __restrict__ src,
                                     const int* __restrict__ dst,
                                     float* __restrict__ out, int nEdges) {
  int t = blockIdx.x * blockDim.x + threadIdx.x;
  int e = t >> 4;
  if (e >= nEdges) return;
  int f = (t & 15) * 4;
  int s = src[e];
  int d = dst[e];
  float nrm = dinv[s] * dinv[d];
  const float4 v = *(const float4*)(x + (size_t)s * 64 + f);
  float* __restrict__ o = out + (size_t)d * 64 + f;
  atomicAdd(o + 0, v.x * nrm);
  atomicAdd(o + 1, v.y * nrm);
  atomicAdd(o + 2, v.z * nrm);
  atomicAdd(o + 3, v.w * nrm);
}

// ---------------------------------------------------------------------------
extern "C" void kernel_launch(void* const* d_in, const int* in_sizes, int n_in,
                              void* d_out, int out_size, void* d_ws, size_t ws_size,
                              hipStream_t stream) {
  const int*   edge_index = (const int*)d_in[0];   // [2, E] int32
  const float* emb        = (const float*)d_in[1]; // [N, 128]
  const float* W1         = (const float*)d_in[2]; // [128, 64]
  const float* b1         = (const float*)d_in[3]; // [64]
  const float* W2         = (const float*)d_in[4]; // [64, 64]
  const float* b2         = (const float*)d_in[5]; // [64]
  float* out = (float*)d_out;                      // [N, 64]

  const int E = in_sizes[0] / 2;
  const int N = in_sizes[1] / 128;
  const int* src = edge_index;
  const int* dst = edge_index + E;

  // Workspace (floats): dinv[N] | x1[N*64] | agg1[N*64] | Bp1 | Bp2.
  // x2 aliases x1 (dead after layer-1 scatter; stream-ordered).
  float* ws   = (float*)d_ws;
  float* dinv = ws;
  size_t npad = ((size_t)N + 63) & ~(size_t)63;
  float* x1   = ws + npad;
  float* agg1 = x1 + (size_t)N * 64;
  float* x2   = x1;
  __bf16* Bp1 = (__bf16*)(agg1 + (size_t)N * 64);  // (128/32)*4*32*16 = 8192 bf16
  __bf16* Bp2 = Bp1 + 8192;                        // (64/32)*4*32*16  = 4096 bf16

  const int tpb = 256;
  const int nBlk = (N + tpb - 1) / tpb;
  const int eBlk = (E + tpb - 1) / tpb;
  const int gemmBlk = (N + 127) / 128;             // 8 waves x 16 rows per block
  const long nf = (long)N * 16;                    // node-feature thread count
  const long ef = (long)E * 16;                    // edge-feature thread count
  const int nfBlk = (int)((nf + tpb - 1) / tpb);
  const int efBlk = (int)((ef + tpb - 1) / tpb);

  // --- weight pre-pack (bf16, WMMA B-register layout) ---
  pack_W_kernel<128><<<2, tpb, 0, stream>>>(W1, Bp1);   // 512 slots
  pack_W_kernel<64><<<1, tpb, 0, stream>>>(W2, Bp2);    // 256 slots

  // --- degrees / D^-1/2 ---
  init_deg_kernel<<<nBlk, tpb, 0, stream>>>(dinv, N);
  count_deg_kernel<<<eBlk, tpb, 0, stream>>>(dst, dinv, E);
  dinv_kernel<<<nBlk, tpb, 0, stream>>>(dinv, N);

  // --- layer 1: x1 = emb @ W1 ; agg1 = selfloop + edge scatter ---
  gemm_wmma_kernel<128, false><<<gemmBlk, 256, 0, stream>>>(emb, Bp1, nullptr, x1, N);
  selfloop_init_kernel<<<nfBlk, tpb, 0, stream>>>(x1, dinv, nullptr, agg1, N);
  scatter_edges_kernel<<<efBlk, tpb, 0, stream>>>(x1, dinv, src, dst, agg1, E);

  // --- layer 2: x2 = relu(agg1 + b1) @ W2 ; out = selfloop + b2 + scatter ---
  gemm_wmma_kernel<64, true><<<gemmBlk, 256, 0, stream>>>(agg1, Bp2, b1, x2, N);
  selfloop_init_kernel<<<nfBlk, tpb, 0, stream>>>(x2, dinv, b2, out, N);
  scatter_edges_kernel<<<efBlk, tpb, 0, stream>>>(x2, dinv, src, dst, out, E);
}
